// Head_65214783422692
// MI455X (gfx1250) — matrix operand order repaired and checked by reading, not compiled, not run on hardware
//
#include <hip/hip_runtime.h>

#define NB 128
#define NT 512
#define NC 464
#define HS 58
#define HP 64    // padded head dim
#define CP 480   // padded embed dim (15 chunks of 32)

typedef __attribute__((ext_vector_type(16))) __bf16 v16bf;
typedef __attribute__((ext_vector_type(8)))  float  v8f;

union V16 {
  v16bf v;
  unsigned short u[16];
  uint4 q[2];
};

__device__ __forceinline__ unsigned short f2bf(float f) {
  unsigned int u = __float_as_uint(f);
  u += 0x7FFFu + ((u >> 16) & 1u);     // round-to-nearest-even
  return (unsigned short)(u >> 16);
}

// ---------------- kernel 1: gather positional bias (T x T) ----------------
__global__ void build_bias(const float* __restrict__ pos_table,
                           const int* __restrict__ pos_idx,
                           float* __restrict__ bias) {
  int i = blockIdx.x * blockDim.x + threadIdx.x;
  if (i < NT * NT) bias[i] = pos_table[pos_idx[i]];
}

// ---------------- kernel 2: repack weights -> bf16 [3][HP][CP] ----------------
// p=0 -> Wq, p=1 -> Wk, p=2 -> Wv; padded rows/cols are zero.
__global__ void repack_w(const float* __restrict__ Wk, const float* __restrict__ Wq,
                         const float* __restrict__ Wv, unsigned short* __restrict__ Wt) {
  int i = blockIdx.x * blockDim.x + threadIdx.x;
  if (i >= 3 * HP * CP) return;
  int p = i / (HP * CP);
  int rem = i % (HP * CP);
  int n = rem / CP;
  int c = rem % CP;
  const float* W = (p == 0) ? Wq : ((p == 1) ? Wk : Wv);
  float val = (n < HS && c < NC) ? W[n * NC + c] : 0.0f;
  Wt[i] = f2bf(val);
}

// load one projection's 4 B-tiles (32x16 each) for K-chunk at ks
__device__ __forceinline__ void load_w4(V16* bm, const unsigned short* __restrict__ Wt,
                                        int p, int li, int hi, int ks) {
#pragma unroll
  for (int nt = 0; nt < 4; ++nt) {
    const unsigned short* wrow =
        Wt + ((size_t)(p * HP + nt * 16 + li)) * CP + ks + 8 * hi;
    bm[nt].q[0] = *(const uint4*)(wrow);
    bm[nt].q[1] = *(const uint4*)(wrow + 16);
  }
}

__device__ __forceinline__ void wmma4(v8f* acc, const V16& a, const V16* bm) {
#pragma unroll
  for (int nt = 0; nt < 4; ++nt) {
    acc[nt] = __builtin_amdgcn_wmma_f32_16x16x32_bf16(
        false, a.v, false, bm[nt].v, (short)0, acc[nt], false, false);
  }
}

// ---------------- kernel 3: fused QKV projection ----------------
// One wave computes one 16-row tile of x against all 3 weight matrices.
// q,k: bf16 [B][T][HP]; v: bf16 transposed [B][HP][T].
__global__ __launch_bounds__(128, 1) void qkv_proj(
    const float* __restrict__ x, const unsigned short* __restrict__ Wt,
    unsigned short* __restrict__ qb, unsigned short* __restrict__ kb,
    unsigned short* __restrict__ vT) {
  const int lane = threadIdx.x & 31;
  const int wave = threadIdx.x >> 5;
  const int tile = blockIdx.x * 4 + wave;          // 0..4095
  const int b  = tile >> 5;
  const int t0 = (tile & 31) << 4;
  const int hi = lane >> 4;
  const int li = lane & 15;

  v8f acc[3][4];
#pragma unroll
  for (int p = 0; p < 3; ++p)
#pragma unroll
    for (int nt = 0; nt < 4; ++nt)
      acc[p][nt] = (v8f){0.f,0.f,0.f,0.f,0.f,0.f,0.f,0.f};

  const float* xrow = x + ((size_t)b * NT + t0 + li) * NC;  // A row m = li

#pragma unroll 1
  for (int kk = 0; kk < 15; ++kk) {
    const int ks = kk * 32;
    // --- B-tile group 0 and 1 loads issued first (stay in flight during A conv)
    V16 b0[4], b1[4], b2[4];
    load_w4(b0, Wt, 0, li, hi, ks);
    load_w4(b1, Wt, 1, li, hi, ks);

    // --- A tile: load x rows, convert f32 -> bf16 (RNE)
    V16 a;
    {
      const int g1 = ks + 8 * hi;                  // elems 0..7 (always in-range)
      float4 f0 = *(const float4*)(xrow + g1);
      float4 f1 = *(const float4*)(xrow + g1 + 4);
      a.u[0] = f2bf(f0.x); a.u[1] = f2bf(f0.y); a.u[2] = f2bf(f0.z); a.u[3] = f2bf(f0.w);
      a.u[4] = f2bf(f1.x); a.u[5] = f2bf(f1.y); a.u[6] = f2bf(f1.z); a.u[7] = f2bf(f1.w);
    }
    {
      const int g2 = ks + 16 + 8 * hi;             // elems 8..15; tail chunk padded
      if (g2 + 7 < NC) {
        float4 f0 = *(const float4*)(xrow + g2);
        float4 f1 = *(const float4*)(xrow + g2 + 4);
        a.u[8]  = f2bf(f0.x); a.u[9]  = f2bf(f0.y); a.u[10] = f2bf(f0.z); a.u[11] = f2bf(f0.w);
        a.u[12] = f2bf(f1.x); a.u[13] = f2bf(f1.y); a.u[14] = f2bf(f1.z); a.u[15] = f2bf(f1.w);
      } else {
#pragma unroll
        for (int i = 8; i < 16; ++i) a.u[i] = 0;
      }
    }

    // --- software pipeline: wmma(p0) overlaps b2 loads; wmma(p1) waits on loads
    // issued a full group earlier; wmma(p2) last.
    wmma4(acc[0], a, b0);
    load_w4(b2, Wt, 2, li, hi, ks);
    wmma4(acc[1], a, b1);
    wmma4(acc[2], a, b2);
  }

  // store: D layout -> lane holds (row t0 + r + 8*hi, col nt*16 + li)
#pragma unroll
  for (int nt = 0; nt < 4; ++nt) {
    const int h = nt * 16 + li;
#pragma unroll
    for (int r = 0; r < 8; ++r) {
      const int t = t0 + r + 8 * hi;
      const size_t o = ((size_t)b * NT + t) * HP + h;
      qb[o] = f2bf(acc[0][nt][r]);
      kb[o] = f2bf(acc[1][nt][r]);
    }
    // v transposed: contiguous in t -> one 16B store
    union { unsigned short u[8]; uint4 q; } st;
#pragma unroll
    for (int r = 0; r < 8; ++r) st.u[r] = f2bf(acc[2][nt][r]);
    *(uint4*)(vT + ((size_t)b * HP + h) * NT + t0 + 8 * hi) = st.q;
  }
}

// ---------------- kernel 4: flash attention ----------------
// One wave per (batch, 16-query tile). Keys processed in causal chunks of 32.
__global__ __launch_bounds__(128, 1) void attn(
    const unsigned short* __restrict__ qb, const unsigned short* __restrict__ kb,
    const unsigned short* __restrict__ vT, const float* __restrict__ bias,
    float* __restrict__ out) {
  __shared__ __align__(32) unsigned short Plds[4][16 * 32];
  const int lane = threadIdx.x & 31;
  const int wave = threadIdx.x >> 5;
  const int tile = blockIdx.x * 4 + wave;          // 0..4095
  const int b  = tile >> 5;
  const int qt = tile & 31;
  const int q0 = qt * 16;
  const int hi = lane >> 4;
  const int li = lane & 15;
  unsigned short* P = Plds[wave];

  // preload Q as two A-chunks (head K = 64)
  V16 aQ[2];
  const unsigned short* qrow = qb + ((size_t)b * NT + q0 + li) * HP;
#pragma unroll
  for (int c = 0; c < 2; ++c) {
    const int g1 = c * 32 + 8 * hi;
    aQ[c].q[0] = *(const uint4*)(qrow + g1);
    aQ[c].q[1] = *(const uint4*)(qrow + g1 + 16);
  }

  float M[8], L[8];
  v8f accO[4];
#pragma unroll
  for (int r = 0; r < 8; ++r) { M[r] = -INFINITY; L[r] = 0.f; }
#pragma unroll
  for (int nt = 0; nt < 4; ++nt)
    accO[nt] = (v8f){0.f,0.f,0.f,0.f,0.f,0.f,0.f,0.f};

  const float scale = 0.13130643285972254f;  // 58^-0.5
  const int nch = (q0 + 16 + 31) / 32;       // causal key chunks of 32

#pragma unroll 1
  for (int j = 0; j < nch; ++j) {
    const int kbase = j * 32;

    // --- issue all independent loads up front so their latency hides behind
    // --- the QK^T WMMAs and the softmax VALU work.
    // K tiles for S = Q (16x64) @ K^T (64x32)
    V16 kB[2][2];
#pragma unroll
    for (int kt = 0; kt < 2; ++kt) {
      const int key = kbase + kt * 16 + li;   // B-matrix N = key index
      const unsigned short* krow = kb + ((size_t)b * NT + key) * HP;
#pragma unroll
      for (int c = 0; c < 2; ++c) {
        kB[kt][c].q[0] = *(const uint4*)(krow + c * 32 + 8 * hi);
        kB[kt][c].q[1] = *(const uint4*)(krow + c * 32 + 8 * hi + 16);
      }
    }
    // positional bias values for this chunk
    float bs0[8], bs1[8];
#pragma unroll
    for (int r = 0; r < 8; ++r) {
      const int q = q0 + r + 8 * hi;
      bs0[r] = bias[q * NT + kbase + li];
      bs1[r] = bias[q * NT + kbase + 16 + li];
    }
    // V tiles for O += P (16x32) @ V (32x64)
    V16 vB[4];
#pragma unroll
    for (int nt = 0; nt < 4; ++nt) {
      const int h = nt * 16 + li;             // B-matrix N = head dim
      const unsigned short* vrow = vT + ((size_t)b * HP + h) * NT + kbase + 8 * hi;
      vB[nt].q[0] = *(const uint4*)(vrow);
      vB[nt].q[1] = *(const uint4*)(vrow + 16);
    }

    // --- QK^T WMMAs
    v8f S[2];
#pragma unroll
    for (int kt = 0; kt < 2; ++kt) {
      v8f sacc = (v8f){0.f,0.f,0.f,0.f,0.f,0.f,0.f,0.f};
#pragma unroll
      for (int c = 0; c < 2; ++c) {
        sacc = __builtin_amdgcn_wmma_f32_16x16x32_bf16(
            false, aQ[c].v, false, kB[kt][c].v, (short)0, sacc, false, false);
      }
      S[kt] = sacc;
    }

    // --- bias + scale + causal mask + online softmax (per row r of this half)
#pragma unroll
    for (int r = 0; r < 8; ++r) {
      const int q = q0 + r + 8 * hi;
      const int key0 = kbase + li;
      const int key1 = kbase + 16 + li;
      float s0 = (S[0][r] + bs0[r]) * scale;
      float s1 = (S[1][r] + bs1[r]) * scale;
      if (key0 > q) s0 = -INFINITY;
      if (key1 > q) s1 = -INFINITY;
      float m = fmaxf(s0, s1);
#pragma unroll
      for (int msk = 1; msk < 16; msk <<= 1) m = fmaxf(m, __shfl_xor(m, msk, 32));
      const float Mn = fmaxf(M[r], m);
      const float cR = __expf(M[r] - Mn);
      const float p0 = __expf(s0 - Mn);
      const float p1 = __expf(s1 - Mn);
      float rs = p0 + p1;
#pragma unroll
      for (int msk = 1; msk < 16; msk <<= 1) rs += __shfl_xor(rs, msk, 32);
      L[r] = L[r] * cR + rs;
      M[r] = Mn;
#pragma unroll
      for (int nt = 0; nt < 4; ++nt) accO[nt][r] = accO[nt][r] * cR;
      // stage P (C/D layout -> LDS) for A-layout reload
      P[(r + 8 * hi) * 32 + li]      = f2bf(p0);
      P[(r + 8 * hi) * 32 + 16 + li] = f2bf(p1);
    }

    asm volatile("s_wait_dscnt 0" ::: "memory");

    // reload P in A layout: m = li, K(key) = (i<8?i:i+8) + 8*hi
    V16 aP;
    aP.q[0] = *(const uint4*)(P + li * 32 + 8 * hi);
    aP.q[1] = *(const uint4*)(P + li * 32 + 16 + 8 * hi);

    // --- PV WMMAs (B operands already resident in vB)
#pragma unroll
    for (int nt = 0; nt < 4; ++nt) {
      accO[nt] = __builtin_amdgcn_wmma_f32_16x16x32_bf16(
          false, aP.v, false, vB[nt].v, (short)0, accO[nt], false, false);
    }
  }

  // epilogue: normalize and store the 58 valid head columns
#pragma unroll
  for (int nt = 0; nt < 4; ++nt) {
    const int h = nt * 16 + li;
    if (h < HS) {
#pragma unroll
      for (int r = 0; r < 8; ++r) {
        const int t = q0 + r + 8 * hi;
        out[((size_t)b * NT + t) * HS + h] = accO[nt][r] / L[r];
      }
    }
  }
}

extern "C" void kernel_launch(void* const* d_in, const int* in_sizes, int n_in,
                              void* d_out, int out_size, void* d_ws, size_t ws_size,
                              hipStream_t stream) {
  const float* x         = (const float*)d_in[0];
  const float* Wk        = (const float*)d_in[1];
  const float* Wq        = (const float*)d_in[2];
  const float* Wv        = (const float*)d_in[3];
  const float* pos_table = (const float*)d_in[4];
  const int*   pos_idx   = (const int*)d_in[5];
  float* out = (float*)d_out;

  char* ws = (char*)d_ws;
  size_t off = 0;
  unsigned short* Wt = (unsigned short*)(ws + off); off += (size_t)3 * HP * CP * 2;   // 184320
  off = (off + 255) & ~(size_t)255;
  unsigned short* qbuf = (unsigned short*)(ws + off); off += (size_t)NB * NT * HP * 2; // 8 MiB
  unsigned short* kbuf = (unsigned short*)(ws + off); off += (size_t)NB * NT * HP * 2; // 8 MiB
  unsigned short* vbuf = (unsigned short*)(ws + off); off += (size_t)NB * HP * NT * 2; // 8 MiB
  float* bias = (float*)(ws + off); off += (size_t)NT * NT * 4;                        // 1 MiB
  (void)ws_size; (void)in_sizes; (void)n_in; (void)out_size;

  repack_w<<<(3 * HP * CP + 255) / 256, 256, 0, stream>>>(Wk, Wq, Wv, Wt);
  build_bias<<<(NT * NT + 255) / 256, 256, 0, stream>>>(pos_table, pos_idx, bias);
  qkv_proj<<<(NB * NT / 16) / 4, 128, 0, stream>>>(x, Wt, qbuf, kbuf, vbuf);
  attn<<<(NB * NT / 16) / 4, 128, 0, stream>>>(qbuf, kbuf, vbuf, bias, out);
}
